// Self_Attention_61907658605336
// MI455X (gfx1250) — matrix operand order repaired and checked by reading, not compiled, not run on hardware
//
#include <hip/hip_runtime.h>

typedef __attribute__((ext_vector_type(16))) _Float16     v16h;
typedef __attribute__((ext_vector_type(8)))  float        v8f;
typedef __attribute__((ext_vector_type(4)))  unsigned int v4u;

#define E_DIM   1024
#define T_DIM   2048
#define N_BATCH 4

static __device__ __forceinline__ v16h frag_from(v4u lo, v4u hi) {
    union { v4u u[2]; v16h h; } t;
    t.u[0] = lo; t.u[1] = hi;
    return t.h;
}

static __device__ __forceinline__ unsigned lds_off(const void* p) {
    // Generic pointers into the LDS aperture carry the LDS byte offset in
    // addr[31:0] (ISA 10.2 aperture mapping), so truncation yields the raw
    // LDS address the async-to-LDS instructions expect.
    return (unsigned)(unsigned long long)p;
}

static __device__ __forceinline__ void async_ld_b128(unsigned lds_addr,
                                                     const void* gptr) {
    asm volatile("global_load_async_to_lds_b128 %0, %1, off"
                 :: "v"(lds_addr), "v"((unsigned long long)gptr)
                 : "memory");
}

static __device__ __forceinline__ void wait_asynccnt0() {
    asm volatile("s_wait_asynccnt 0x0" ::: "memory");
}

// ---------------------------------------------------------------------------
// Generic WMMA GEMM:  C[M,N] = A[M,K] * Bt[N,K]^T  (+ bias[N])
//   A  : f16 row-major, leading dim lda
//   Bt : f16, stored as B-transposed (row n holds column n of B), leading dim ldb
//   mode 0: C f16 row-major   C[m*ldc + n]
//   mode 1: C f16 transposed  C[n*ldc + m]
//   mode 2: C f32 row-major   C[m*ldc + n]
// Block: 256 threads (8 waves). Block tile 128x128, wave tile 32x64.
// Double-buffered LDS staging via global_load_async_to_lds_b128 (ASYNCcnt).
// Grid: (N/128, M/128, batches). All dims are exact multiples of 128 here.
// ---------------------------------------------------------------------------
__global__ __launch_bounds__(256)
void wmma_gemm_kernel(const _Float16* __restrict__ A,  long sA, int lda,
                      const _Float16* __restrict__ Bt, long sB, int ldb,
                      const float*    __restrict__ bias,
                      void* __restrict__ Cv, long sC, int ldc,
                      int K, int mode)
{
    // 2 buffers x (A panel 128x32 f16 = 8KB, B panel 128x32 f16 = 8KB)
    __shared__ __align__(16) char smem[2 * 16384];

    const int tid  = threadIdx.x;
    const int lane = tid & 31;
    const int wave = tid >> 5;
    const int half = lane >> 4;      // which 16-lane half of the wave
    const int l16  = lane & 15;
    const int wm   = wave >> 1;      // 0..3 : wave row within block tile
    const int wn   = wave & 1;       // 0..1 : wave col within block tile

    const int rowBase0 = blockIdx.y * 128;   // block's first A row
    const int colBase0 = blockIdx.x * 128;   // block's first Bt row (C col)
    const int z        = blockIdx.z;

    A  += (long)z * sA;
    Bt += (long)z * sB;

    const unsigned smemBase = lds_off(smem);

    // Issue async copies for one 128x32 k-panel of A and of Bt into buffer buf.
    // 512 16B chunks per panel, 2 per thread per panel (4 asyncs/thread).
    auto issue_tile = [&](int kb, int buf) {
        const unsigned baseA = smemBase + buf * 16384;
        const unsigned baseB = baseA + 8192;
#pragma unroll
        for (int s = 0; s < 2; ++s) {
            const int c   = tid + s * 256;
            const int row = c >> 2;
            const int ch  = c & 3;
            async_ld_b128(baseA + row * 64 + ch * 16,
                          A + (long)(rowBase0 + row) * lda + kb + ch * 8);
        }
#pragma unroll
        for (int s = 0; s < 2; ++s) {
            const int c   = tid + s * 256;
            const int row = c >> 2;
            const int ch  = c & 3;
            async_ld_b128(baseB + row * 64 + ch * 16,
                          Bt + (long)(colBase0 + row) * ldb + kb + ch * 8);
        }
    };

    v8f acc[2][4];
#pragma unroll
    for (int i = 0; i < 2; ++i)
#pragma unroll
        for (int j = 0; j < 4; ++j)
            acc[i][j] = (v8f){0.f,0.f,0.f,0.f,0.f,0.f,0.f,0.f};

    issue_tile(0, 0);

    for (int kb = 0; kb < K; kb += 32) {
        const int buf = (kb >> 5) & 1;

        wait_asynccnt0();       // this wave's async copies have landed
        __syncthreads();        // everyone's copies have landed; prev reads done

        if (kb + 32 < K) issue_tile(kb + 32, buf ^ 1);

        const char* sAp = smem + buf * 16384;
        const char* sBp = sAp + 8192;

        v16h a[2], b[4];
        // A fragment (16x32 f16): elems 0..7 -> K = half*8+0..7,
        // elems 8..15 -> K = 16 + half*8 + 0..7  (ISA 7.12.2). Row pitch 64B.
#pragma unroll
        for (int i = 0; i < 2; ++i) {
            const char* rp = sAp + (wm * 32 + i * 16 + l16) * 64;
            v4u lo = *reinterpret_cast<const v4u*>(rp + half * 16);
            v4u hi = *reinterpret_cast<const v4u*>(rp + 32 + half * 16);
            a[i] = frag_from(lo, hi);
        }
        // B fragment (32x16 f16): elems 0..7 -> K = half*8+0..7, elems 8..15 -> K = 16 + half*8 + 0..7.
#pragma unroll
        for (int j = 0; j < 4; ++j) {
            const char* rp = sBp + (wn * 64 + j * 16 + l16) * 64;
            v4u lo = *reinterpret_cast<const v4u*>(rp + half * 16);
            v4u hi = *reinterpret_cast<const v4u*>(rp + 32 + half * 16);
            b[j] = frag_from(lo, hi);
        }
#pragma unroll
        for (int i = 0; i < 2; ++i)
#pragma unroll
            for (int j = 0; j < 4; ++j)
                acc[i][j] = __builtin_amdgcn_wmma_f32_16x16x32_f16(
                    false, a[i], false, b[j], (short)0, acc[i][j], false, false);
    }

    // Epilogue. C/D layout: VGPR r, lane -> (m = r + half*8, n = l16) per 16x16.
    const int rbase = rowBase0 + wm * 32;
    const int cbase = colBase0 + wn * 64;

    if (mode == 2) {
        float* C = (float*)Cv + (long)z * sC;
#pragma unroll
        for (int i = 0; i < 2; ++i)
#pragma unroll
            for (int j = 0; j < 4; ++j) {
                const int col = cbase + j * 16 + l16;
                const float bv = bias ? bias[col] : 0.f;
#pragma unroll
                for (int r = 0; r < 8; ++r) {
                    const int m = rbase + i * 16 + half * 8 + r;
                    C[(long)m * ldc + col] = acc[i][j][r] + bv;
                }
            }
    } else if (mode == 1) {
        _Float16* C = (_Float16*)Cv + (long)z * sC;
#pragma unroll
        for (int i = 0; i < 2; ++i)
#pragma unroll
            for (int j = 0; j < 4; ++j) {
                const int col = cbase + j * 16 + l16;
                const float bv = bias ? bias[col] : 0.f;
#pragma unroll
                for (int r = 0; r < 8; ++r) {
                    const int m = rbase + i * 16 + half * 8 + r;
                    C[(long)col * ldc + m] = (_Float16)(acc[i][j][r] + bv);
                }
            }
    } else {
        _Float16* C = (_Float16*)Cv + (long)z * sC;
#pragma unroll
        for (int i = 0; i < 2; ++i)
#pragma unroll
            for (int j = 0; j < 4; ++j) {
                const int col = cbase + j * 16 + l16;
                const float bv = bias ? bias[col] : 0.f;
#pragma unroll
                for (int r = 0; r < 8; ++r) {
                    const int m = rbase + i * 16 + half * 8 + r;
                    C[(long)m * ldc + col] = (_Float16)(acc[i][j][r] + bv);
                }
            }
    }
}

// ---------------------------------------------------------------------------
// In-place row softmax over f16 rows (row length = row_len), f32 math.
// One block (256 threads) per row.
// ---------------------------------------------------------------------------
__global__ __launch_bounds__(256)
void softmax_rows_f16(_Float16* __restrict__ S, int row_len)
{
    __shared__ float red[256];
    const int tid = threadIdx.x;
    _Float16* p = S + (long)blockIdx.x * row_len;

    float mx = -3.4e38f;
    for (int i = tid; i < row_len; i += 256) mx = fmaxf(mx, (float)p[i]);
    red[tid] = mx; __syncthreads();
    for (int s = 128; s > 0; s >>= 1) {
        if (tid < s) red[tid] = fmaxf(red[tid], red[tid + s]);
        __syncthreads();
    }
    mx = red[0]; __syncthreads();

    float sum = 0.f;
    for (int i = tid; i < row_len; i += 256) sum += __expf((float)p[i] - mx);
    red[tid] = sum; __syncthreads();
    for (int s = 128; s > 0; s >>= 1) {
        if (tid < s) red[tid] += red[tid + s];
        __syncthreads();
    }
    const float inv = 1.0f / red[0];

    for (int i = tid; i < row_len; i += 256)
        p[i] = (_Float16)(__expf((float)p[i] - mx) * inv);
}

// ---------------------------------------------------------------------------
// Transpose + convert the three ExE weight matrices: Wt[f][e] = (f16)W[e][f]
// ---------------------------------------------------------------------------
__global__ __launch_bounds__(256)
void transpose_convert_w(const float* __restrict__ W0, const float* __restrict__ W1,
                         const float* __restrict__ W2,
                         _Float16* __restrict__ O0, _Float16* __restrict__ O1,
                         _Float16* __restrict__ O2)
{
    __shared__ float tile[32][33];
    const float* W = (blockIdx.z == 0) ? W0 : (blockIdx.z == 1) ? W1 : W2;
    _Float16*   O = (blockIdx.z == 0) ? O0 : (blockIdx.z == 1) ? O1 : O2;

    const int x  = blockIdx.x * 32 + threadIdx.x;
    const int y0 = blockIdx.y * 32;
    for (int dy = threadIdx.y; dy < 32; dy += 8)
        tile[dy][threadIdx.x] = W[(long)(y0 + dy) * E_DIM + x];
    __syncthreads();

    const int xo  = blockIdx.y * 32 + threadIdx.x;
    const int yo0 = blockIdx.x * 32;
    for (int dy = threadIdx.y; dy < 32; dy += 8)
        O[(long)(yo0 + dy) * E_DIM + xo] = (_Float16)tile[threadIdx.x][dy];
}

// ---------------------------------------------------------------------------
// fp32 -> f16 elementwise convert (for X)
// ---------------------------------------------------------------------------
__global__ __launch_bounds__(256)
void convert_f32_f16(const float* __restrict__ in, _Float16* __restrict__ out, long n)
{
    long i = (long)blockIdx.x * blockDim.x + threadIdx.x;
    const long stride = (long)gridDim.x * blockDim.x;
    for (; i < n; i += stride) out[i] = (_Float16)in[i];
}

extern "C" void kernel_launch(void* const* d_in, const int* in_sizes, int n_in,
                              void* d_out, int out_size, void* d_ws, size_t ws_size,
                              hipStream_t stream)
{
    const float* X  = (const float*)d_in[0];
    const float* Wq = (const float*)d_in[1];
    const float* bq = (const float*)d_in[2];
    const float* Wk = (const float*)d_in[3];
    const float* bk = (const float*)d_in[4];
    const float* Wv = (const float*)d_in[5];
    const float* bv = (const float*)d_in[6];
    float* out = (float*)d_out;

    const long MT = (long)N_BATCH * T_DIM;   // 8192 token rows
    char* ws = (char*)d_ws;
    _Float16* Xh  = (_Float16*)ws;  ws += MT * E_DIM * 2;                 // 16 MiB
    _Float16* Wtq = (_Float16*)ws;  ws += (long)E_DIM * E_DIM * 2;        //  2 MiB
    _Float16* Wtk = (_Float16*)ws;  ws += (long)E_DIM * E_DIM * 2;
    _Float16* Wtv = (_Float16*)ws;  ws += (long)E_DIM * E_DIM * 2;
    _Float16* Qh  = (_Float16*)ws;  ws += MT * E_DIM * 2;                 // 16 MiB
    _Float16* Kh  = (_Float16*)ws;  ws += MT * E_DIM * 2;                 // 16 MiB
    _Float16* Vt  = (_Float16*)ws;  ws += (long)E_DIM * MT * 2;          // 16 MiB, Vt[e][token]
    _Float16* S   = (_Float16*)ws;  ws += (long)N_BATCH * T_DIM * T_DIM * 2; // 32 MiB

    // 1) convert X to f16
    convert_f32_f16<<<4096, 256, 0, stream>>>(X, Xh, MT * E_DIM);

    // 2) transpose + convert weights
    transpose_convert_w<<<dim3(E_DIM/32, E_DIM/32, 3), dim3(32, 8), 0, stream>>>(
        Wq, Wk, Wv, Wtq, Wtk, Wtv);

    // 3) QKV projections: [8192,1024] = Xh @ W (+bias)
    const dim3 gProj(E_DIM/128, MT/128, 1);
    wmma_gemm_kernel<<<gProj, 256, 0, stream>>>(Xh, 0, E_DIM, Wtq, 0, E_DIM, bq,
                                                Qh, 0, E_DIM, E_DIM, /*mode=*/0);
    wmma_gemm_kernel<<<gProj, 256, 0, stream>>>(Xh, 0, E_DIM, Wtk, 0, E_DIM, bk,
                                                Kh, 0, E_DIM, E_DIM, /*mode=*/0);
    // V written transposed: Vt[e][token]  (ldc = MT)
    wmma_gemm_kernel<<<gProj, 256, 0, stream>>>(Xh, 0, E_DIM, Wtv, 0, E_DIM, bv,
                                                Vt, 0, (int)MT, E_DIM, /*mode=*/1);

    // 4) sim[n] = Q[n] @ K[n]^T  -> S f16 [N,T,T]
    const dim3 gSim(T_DIM/128, T_DIM/128, N_BATCH);
    wmma_gemm_kernel<<<gSim, 256, 0, stream>>>(Qh, (long)T_DIM * E_DIM, E_DIM,
                                               Kh, (long)T_DIM * E_DIM, E_DIM,
                                               nullptr,
                                               S, (long)T_DIM * T_DIM, T_DIM,
                                               E_DIM, /*mode=*/0);

    // 5) row softmax over S (in place, f16)
    softmax_rows_f16<<<(int)MT, 256, 0, stream>>>(S, T_DIM);

    // 6) out[n] = P[n] @ V[n]   (Bt = Vt, batch offset is T_DIM columns)
    const dim3 gOut(E_DIM/128, T_DIM/128, N_BATCH);
    wmma_gemm_kernel<<<gOut, 256, 0, stream>>>(S, (long)T_DIM * T_DIM, T_DIM,
                                               Vt, (long)T_DIM, (int)MT,
                                               nullptr,
                                               out, (long)T_DIM * E_DIM, E_DIM,
                                               T_DIM, /*mode=*/2);
}
